// JointForceRestoring_35046933135563
// MI455X (gfx1250) — matrix operand order repaired and checked by reading, not compiled, not run on hardware
//
#include <hip/hip_runtime.h>
#include <math.h>

// ---- problem constants (match reference) ----
#define Bsz   64
#define Cch   256
#define Tt    128
#define Vv    25
#define VRr   25
#define NINT  128
#define BN_EPS 1e-5f

// ---- CDNA5 WMMA types (trivial ext-vector types; union-safe) ----
typedef __attribute__((ext_vector_type(16))) __bf16        v16bf;
typedef __attribute__((ext_vector_type(8)))  float         v8f;
typedef __attribute__((ext_vector_type(4)))  unsigned int  u32x4;
typedef __attribute__((ext_vector_type(4)))  float         f32x4;

// workspace layout (floats): [0,256)=tw  [256,512)=pw  [512]=a0 [513]=b0
// [768,1024)=bn_scale  [1024,1280)=bn_shift ; byte offset 8192: gwbf16[256*256]

union BfBits { __bf16 h; unsigned short u; };
union FragU  { u32x4 q[2]; v16bf v; };

// ---------------------------------------------------------------------------
// K0: fold theta/phi projections into per-channel vectors; convert g_w -> bf16
// ---------------------------------------------------------------------------
__global__ __launch_bounds__(256) void prep_kernel(
    const float* __restrict__ theta_w, const float* __restrict__ theta_b,
    const float* __restrict__ phi_w,   const float* __restrict__ phi_b,
    const float* __restrict__ w1,      const float* __restrict__ w2,
    const float* __restrict__ g_w,
    float* __restrict__ wsf, unsigned short* __restrict__ gwbf)
{
    const int tid = threadIdx.x;
    float tw = 0.f, pw = 0.f;
    for (int i = 0; i < NINT; ++i) {
        tw += w1[i] * theta_w[i * Cch + tid];
        pw += w2[i] * phi_w[i * Cch + tid];
    }
    wsf[tid]       = tw;
    wsf[256 + tid] = pw;
    if (tid == 0) {
        float a0 = 0.f, b0 = 0.f;
        for (int i = 0; i < NINT; ++i) { a0 += w1[i] * theta_b[i]; b0 += w2[i] * phi_b[i]; }
        wsf[512] = a0; wsf[513] = b0;
    }
    // g_w is [g][c] row-major == the [N][K] layout WMMA B-fragments want.
    for (int j = tid; j < Cch * Cch; j += 256) {
        BfBits bb; bb.h = (__bf16)g_w[j];
        gwbf[j] = bb.u;
    }
}

// ---------------------------------------------------------------------------
// K1: fused per-(b,t) block: a/b dots, softmax attention, m = att@y_,
//     out = m @ g_w^T via v_wmma_f32_16x16x32_bf16 (B from L2, A from LDS)
// ---------------------------------------------------------------------------
__global__ __launch_bounds__(256) void fused_kernel(
    const float* __restrict__ x, const float* __restrict__ y,
    const float* __restrict__ g_b, const float* __restrict__ Ck,
    const float* __restrict__ wsf, const unsigned short* __restrict__ gwbf_u,
    float* __restrict__ zout)
{
    __shared__ float  ysh[Cch * Vv];     // y tile [c][w], f32        (25.6 KB)
    __shared__ __bf16 msh[32 * Cch];     // A matrix [v][c], bf16     (16 KB)
    __shared__ float  attsh[32 * 26];    // attention rows (padded)
    __shared__ float  af[32], bfA[32], svsh[32];

    const int tid = threadIdx.x;
    const int t = blockIdx.x, b = blockIdx.y;

    const float twc = wsf[tid], pwc = wsf[256 + tid];
    const float a0  = wsf[512], b0  = wsf[513];

    if (tid < 32) { af[tid] = a0; bfA[tid] = b0; svsh[tid] = 0.f; }
    __syncthreads();

    // ---- stage A: stream x/y channel rows; wave-reduce dot products ----
    {
        const int base = ((b * Cch + tid) * Tt + t) * Vv;  // same for x and y
        #pragma unroll
        for (int v = 0; v < Vv; ++v) {
            float px = x[base + v] * twc;
            #pragma unroll
            for (int o = 16; o > 0; o >>= 1) px += __shfl_down(px, o, 32);
            if ((tid & 31) == 0) atomicAdd(&af[v], px);
        }
        #pragma unroll
        for (int w = 0; w < VRr; ++w) {
            float yv = y[base + w];
            ysh[tid * Vv + w] = yv;
            float py = yv * pwc;
            #pragma unroll
            for (int o = 16; o > 0; o >>= 1) py += __shfl_down(py, o, 32);
            if ((tid & 31) == 0) atomicAdd(&bfA[w], py);
        }
    }
    __syncthreads();

    // ---- stage B: leaky-relu + softmax + C_k bias, row-sum s_v for g_b ----
    if (tid < Vv) {
        const int v = tid;
        const float av = af[v];
        float fr[VRr], mx = -1e30f;
        #pragma unroll
        for (int w = 0; w < VRr; ++w) {
            float f = av + bfA[w];
            f = (f < 0.f) ? 0.2f * f : f;
            fr[w] = f; mx = fmaxf(mx, f);
        }
        float s = 0.f;
        #pragma unroll
        for (int w = 0; w < VRr; ++w) { fr[w] = __expf(fr[w] - mx); s += fr[w]; }
        const float inv = 1.f / s;
        float sv = 0.f;
        #pragma unroll
        for (int w = 0; w < VRr; ++w) {
            float at = fr[w] * inv + Ck[v * VRr + w];
            attsh[v * 26 + w] = at;
            sv += at;
        }
        svsh[v] = sv;
    }
    __syncthreads();

    // ---- stage C: m[v][c] = sum_w att[v][w] * y[c][w]  -> bf16 A matrix ----
    {
        float yr[VRr];
        #pragma unroll
        for (int w = 0; w < VRr; ++w) yr[w] = ysh[tid * Vv + w];
        for (int v = 0; v < 32; ++v) {
            float acc = 0.f;
            if (v < Vv) {
                #pragma unroll
                for (int w = 0; w < VRr; ++w) acc += attsh[v * 26 + w] * yr[w];
            }
            msh[v * Cch + tid] = (__bf16)acc;
        }
    }
    __syncthreads();

    // ---- stage D: out[v][g] = m @ g_w^T via WMMA; 2x2 tiles per wave ----
    {
        const int wv   = tid >> 5;
        const int lane = tid & 31;
        const int hsel = lane >> 4;     // 0: lanes 0-15, 1: lanes 16-31
        const int l16  = lane & 15;
        const int nb   = wv * 32;       // this wave's 32 output channels
        const u32x4* gw4 = (const u32x4*)gwbf_u;

        v8f acc00 = {}; v8f acc01 = {}; v8f acc10 = {}; v8f acc11 = {};

        for (int k0 = 0; k0 < Cch; k0 += 32) {
            FragU A0, A1, B0, B1;
            // A fragments (LDS): lanes 0-15 take K k0+0..7 / k0+16..23,
            //                    lanes 16-31 take K k0+8..15 / k0+24..31
            const int ka = k0 + hsel * 8;
            const __bf16* ar0 = &msh[(0  + l16) * Cch + ka];
            const __bf16* ar1 = &msh[(16 + l16) * Cch + ka];
            A0.q[0] = *(const u32x4*)(ar0);
            A0.q[1] = *(const u32x4*)(ar0 + 16);
            A1.q[0] = *(const u32x4*)(ar1);
            A1.q[1] = *(const u32x4*)(ar1 + 16);
            // B fragments (global, L2-resident g_w bf16 in [N][K] layout):
            // lanes 0-15: K k0+0..7 / k0+16..23; lanes 16-31: K k0+8..15 / k0+24..31
            const int kb = k0 + hsel * 8;
            const int o0 = ((nb      + l16) * Cch + kb) >> 3;  // u32x4 index
            const int o1 = ((nb + 16 + l16) * Cch + kb) >> 3;
            B0.q[0] = gw4[o0]; B0.q[1] = gw4[o0 + 2];
            B1.q[0] = gw4[o1]; B1.q[1] = gw4[o1 + 2];

            acc00 = __builtin_amdgcn_wmma_f32_16x16x32_bf16(false, A0.v, false, B0.v, (short)0, acc00, false, false);
            acc01 = __builtin_amdgcn_wmma_f32_16x16x32_bf16(false, A0.v, false, B1.v, (short)0, acc01, false, false);
            acc10 = __builtin_amdgcn_wmma_f32_16x16x32_bf16(false, A1.v, false, B0.v, (short)0, acc10, false, false);
            acc11 = __builtin_amdgcn_wmma_f32_16x16x32_bf16(false, A1.v, false, B1.v, (short)0, acc11, false, false);
        }

        // epilogue: add s_v * g_b[g], store z in [B,C,T,V] layout
        const int zbase = b * Cch * Tt * Vv + t * Vv;
        auto store_tile = [&](v8f acc, int m0, int n0) {
            const int g = n0 + l16;
            const float gb = g_b[g];
            const int base = zbase + g * (Tt * Vv);
            const int vb = m0 + hsel * 8;
            #pragma unroll
            for (int r = 0; r < 8; ++r) {
                const int v = vb + r;
                if (v < Vv) zout[base + v] = acc[r] + svsh[v] * gb;
            }
        };
        store_tile(acc00, 0,  nb);
        store_tile(acc01, 0,  nb + 16);
        store_tile(acc10, 16, nb);
        store_tile(acc11, 16, nb + 16);
    }
}

// ---------------------------------------------------------------------------
// K2: per-channel batch statistics -> scale/shift
// ---------------------------------------------------------------------------
__global__ __launch_bounds__(256) void bnstats_kernel(
    const float* __restrict__ z, const float* __restrict__ gamma,
    const float* __restrict__ beta, float* __restrict__ wsf)
{
    __shared__ float ssum[256], ssq[256];
    const int g = blockIdx.x, tid = threadIdx.x;
    float s = 0.f, q = 0.f;
    for (int b = 0; b < Bsz; ++b) {
        const float* p = z + (b * Cch + g) * (Tt * Vv);
        for (int i = tid; i < Tt * Vv; i += 256) { float v = p[i]; s += v; q += v * v; }
    }
    ssum[tid] = s; ssq[tid] = q; __syncthreads();
    for (int o = 128; o > 0; o >>= 1) {
        if (tid < o) { ssum[tid] += ssum[tid + o]; ssq[tid] += ssq[tid + o]; }
        __syncthreads();
    }
    if (tid == 0) {
        const float n    = (float)(Bsz * Tt * Vv);
        const float mean = ssum[0] / n;
        const float var  = ssq[0] / n - mean * mean;
        const float sc   = gamma[g] * rsqrtf(var + BN_EPS);
        wsf[768 + g]  = sc;
        wsf[1024 + g] = beta[g] - mean * sc;
    }
}

// ---------------------------------------------------------------------------
// K3: in-place normalize + ReLU over d_out (float4 per thread)
// ---------------------------------------------------------------------------
__global__ __launch_bounds__(256) void bnapply_kernel(
    float* __restrict__ z, const float* __restrict__ wsf)
{
    const int i = blockIdx.x * 256 + threadIdx.x;         // float4 index
    const int g = (i / 800) & 255;                        // 3200 floats per (b,g)
    const float sc = wsf[768 + g], sh = wsf[1024 + g];
    f32x4 v = ((const f32x4*)z)[i];
    v.x = fmaxf(v.x * sc + sh, 0.f);
    v.y = fmaxf(v.y * sc + sh, 0.f);
    v.z = fmaxf(v.z * sc + sh, 0.f);
    v.w = fmaxf(v.w * sc + sh, 0.f);
    ((f32x4*)z)[i] = v;
}

// ---------------------------------------------------------------------------
extern "C" void kernel_launch(void* const* d_in, const int* in_sizes, int n_in,
                              void* d_out, int out_size, void* d_ws, size_t ws_size,
                              hipStream_t stream)
{
    const float* x       = (const float*)d_in[0];
    const float* y       = (const float*)d_in[1];
    const float* g_w     = (const float*)d_in[2];
    const float* g_b     = (const float*)d_in[3];
    const float* theta_w = (const float*)d_in[4];
    const float* theta_b = (const float*)d_in[5];
    const float* phi_w   = (const float*)d_in[6];
    const float* phi_b   = (const float*)d_in[7];
    const float* w1      = (const float*)d_in[8];
    const float* w2      = (const float*)d_in[9];
    const float* Ck      = (const float*)d_in[10];
    const float* gamma   = (const float*)d_in[11];
    const float* beta    = (const float*)d_in[12];

    float*          out  = (float*)d_out;
    float*          wsf  = (float*)d_ws;
    unsigned short* gwbf = (unsigned short*)((char*)d_ws + 8192);   // 128 KB bf16 g_w

    prep_kernel<<<1, 256, 0, stream>>>(theta_w, theta_b, phi_w, phi_b, w1, w2,
                                       g_w, wsf, gwbf);
    fused_kernel<<<dim3(Tt, Bsz), 256, 0, stream>>>(x, y, g_b, Ck, wsf, gwbf, out);
    bnstats_kernel<<<Cch, 256, 0, stream>>>(out, gamma, beta, wsf);
    bnapply_kernel<<<(Bsz * Cch * Tt * Vv) / (4 * 256), 256, 0, stream>>>(out, wsf);
}
